// TypedHGNNConv_41566693491333
// MI455X (gfx1250) — compile-verified, matrix operand order checked by reading, not run on hardware
//
#include <hip/hip_runtime.h>
#include <hip/hip_bf16.h>
#include <math.h>

// ---------------- problem constants (from reference setup_inputs) -----------
#define R_  4
#define N_  10000
#define E_  2000
#define F_  128
#define O_  128
#define EPS_ 1e-8f

// LDS tile row stride (bf16 elements); %8==0 -> rows are 16B aligned
#define LDK 40

typedef __attribute__((ext_vector_type(16))) __bf16 v16bf;
typedef __attribute__((ext_vector_type(8)))  float  v8f;
typedef __attribute__((ext_vector_type(4)))  float  v4f;   // native vector for NT loads

__device__ __forceinline__ int imin(int a, int b) { return a < b ? a : b; }

union FragU { uint4 q[2]; v16bf v; };

// Load a 16x32 bf16 WMMA operand fragment from LDS.
// LDS layout: 16 rows (M for A / N for B) x 32 K values, row stride LDK bf16.
// Documented 16-bit A layout (ISA 7.12.2):
//   lanes 0-15 : M=lane,    elements {K0..7,  K16..23}
//   lanes 16-31: M=lane-16, elements {K8..15, K24..31}
__device__ __forceinline__ v16bf lds_load_frag(const __bf16* base) {
  const int lane = threadIdx.x & 31;
  const int h = (lane >> 4) & 1;
  const uint4* p = reinterpret_cast<const uint4*>(base + (size_t)(lane & 15) * LDK);
  FragU u;
  u.q[0] = p[h];
  u.q[1] = p[2 + h];
  return u.v;
}

// pack two f32 into two bf16 in one dword (v_cvt_pk_bf16_f32)
__device__ __forceinline__ unsigned pk2(float lo, float hi) {
  union { __bf16 h[2]; unsigned u; } p;
  p.h[0] = (__bf16)lo;
  p.h[1] = (__bf16)hi;
  return p.u;
}

// Store a 4x4 f32 micro-tile transposed into LDS as bf16 (4x ds_store_b64).
__device__ __forceinline__ void commit_tr4f(__bf16* dst, int c0, int r0,
                                            const float ar[4][4]) {
#pragma unroll
  for (int i = 0; i < 4; ++i) {
    uint2 d;
    d.x = pk2(ar[0][i], ar[1][i]);
    d.y = pk2(ar[2][i], ar[3][i]);
    *reinterpret_cast<uint2*>(dst + (c0 + i) * LDK + r0) = d;
  }
}

// Store a 4x4 bf16 micro-tile (rows held as uint2) transposed into LDS.
__device__ __forceinline__ void commit_tr4(__bf16* dst, int c0, int r0,
                                           const uint2 br[4]) {
  unsigned s[4][4];
#pragma unroll
  for (int j = 0; j < 4; ++j) {
    s[j][0] = br[j].x & 0xffffu;
    s[j][1] = br[j].x >> 16;
    s[j][2] = br[j].y & 0xffffu;
    s[j][3] = br[j].y >> 16;
  }
#pragma unroll
  for (int i = 0; i < 4; ++i) {
    uint2 d;
    d.x = s[0][i] | (s[1][i] << 16);
    d.y = s[2][i] | (s[3][i] << 16);
    *reinterpret_cast<uint2*>(dst + (c0 + i) * LDK + r0) = d;
  }
}

// ---------------- degree kernels -------------------------------------------
__global__ void k_dv(const float* __restrict__ H, float* __restrict__ dv) {
  const int r = blockIdx.y;
  const int w = threadIdx.x >> 5;
  const int lane = threadIdx.x & 31;
  const int n = blockIdx.x * 8 + w;
  if (n >= N_) return;
  const float4* row = reinterpret_cast<const float4*>(H + ((size_t)r * N_ + n) * E_);
  float s = 0.f;
  for (int i = lane; i < E_ / 4; i += 32) {
    float4 v = row[i];
    s += v.x + v.y + v.z + v.w;
  }
#pragma unroll
  for (int off = 16; off > 0; off >>= 1) s += __shfl_xor(s, off, 32);
  if (lane == 0) dv[(size_t)r * N_ + n] = rsqrtf(s + EPS_);
}

__global__ void k_de_part(const float* __restrict__ H, float* __restrict__ de) {
  const int r = blockIdx.z;
  const int e = blockIdx.x * 256 + threadIdx.x;
  if (e >= E_) return;
  const int n0 = blockIdx.y * 250;
  const float* base = H + (size_t)r * N_ * E_ + (size_t)n0 * E_ + e;
  float s = 0.f;
  for (int i = 0; i < 250; ++i) s += base[(size_t)i * E_];
  atomicAdd(&de[(size_t)r * E_ + e], s);
}

__global__ void k_de_fin(float* __restrict__ de) {
  int i = blockIdx.x * 256 + threadIdx.x;
  if (i < R_ * E_) de[i] = rsqrtf(de[i] + EPS_);
}

__global__ void k_scaleX(const float* __restrict__ X, const float* __restrict__ dv,
                         __bf16* __restrict__ Xs) {
  size_t idx = (size_t)blockIdx.x * 256 + threadIdx.x;
  if (idx >= (size_t)R_ * N_ * F_) return;
  int f = (int)(idx % F_);
  size_t t = idx / F_;
  int n = (int)(t % N_);
  int r = (int)(t / N_);
  Xs[idx] = (__bf16)(dv[(size_t)r * N_ + n] * X[(size_t)n * F_ + f]);
}

// ---------------- GEMM1: tX[r,e,f] = de_is[r,e] * sum_n H[r,n,e]*Xs[r,n,f] --
__global__ void __launch_bounds__(256)
k_gemm1(const float* __restrict__ H, const __bf16* __restrict__ Xs,
        const float* __restrict__ de, __bf16* __restrict__ tX) {
  __shared__ alignas(16) __bf16 Abuf[2][128 * LDK];   // [e_local][n_local]
  __shared__ alignas(16) __bf16 Bbuf[2][128 * LDK];   // [f][n_local]
  const int r  = blockIdx.y;
  const int e0 = blockIdx.x * 128;
  const int t  = threadIdx.x;
  const int w  = t >> 5;
  const int lane = t & 31;
  const int r0 = (t >> 5) * 4;   // n_local group 0..28
  const int c0 = (t & 31) * 4;   // column group  0..124

  const size_t Hr = (size_t)r * N_ * E_;
  const size_t Xr = (size_t)r * N_ * F_;
  const bool e_full = (e0 + 128 <= E_);

  v8f acc[8];
  v8f zero = {};
#pragma unroll
  for (int i = 0; i < 8; ++i) acc[i] = zero;

  float Ar[4][4];   // H micro-tile: [row n][col e]
  uint2 Br[4];      // Xs rows: 4 bf16 each

  auto stage = [&](int n0) {
    const bool n_full = (n0 + 32 <= N_);
    if (n_full && e_full) {
#pragma unroll
      for (int j = 0; j < 4; ++j) {
        const v4f* rp = reinterpret_cast<const v4f*>(
            H + Hr + (size_t)(n0 + r0 + j) * E_ + (e0 + c0));
        v4f v = __builtin_nontemporal_load(rp);   // H is single-use: NT
        Ar[j][0] = v[0]; Ar[j][1] = v[1]; Ar[j][2] = v[2]; Ar[j][3] = v[3];
      }
#pragma unroll
      for (int j = 0; j < 4; ++j)
        Br[j] = *reinterpret_cast<const uint2*>(
            Xs + Xr + (size_t)(n0 + r0 + j) * F_ + c0);
    } else {
      // branchless edge path: clamped addresses + selects (no exec juggling)
#pragma unroll
      for (int j = 0; j < 4; ++j) {
        const int n = n0 + r0 + j;
        const bool nok = n < N_;
        const size_t ro = Hr + (size_t)imin(n, N_ - 1) * E_;
#pragma unroll
        for (int i = 0; i < 4; ++i) {
          const int e = e0 + c0 + i;
          float v = H[ro + imin(e, E_ - 1)];
          Ar[j][i] = (nok && e < E_) ? v : 0.f;
        }
        uint2 b = *reinterpret_cast<const uint2*>(
            Xs + Xr + (size_t)imin(n, N_ - 1) * F_ + c0);
        if (!nok) { b.x = 0u; b.y = 0u; }
        Br[j] = b;
      }
    }
  };
  auto commit = [&](int buf) {
    commit_tr4f(&Abuf[buf][0], c0, r0, Ar);
    commit_tr4 (&Bbuf[buf][0], c0, r0, Br);
  };

  const int STEPS = (N_ + 31) / 32;   // 313
  stage(0);
  commit(0);
  for (int it = 0; it < STEPS; ++it) {
    const bool more = (it + 1) < STEPS;
    if (more) stage((it + 1) * 32);   // global loads for next tile, pre-barrier
    __syncthreads();
    const __bf16* Ab = &Abuf[it & 1][0];
    const __bf16* Bb = &Bbuf[it & 1][0];
    v16bf a = lds_load_frag(Ab + (w * 16) * LDK);
#pragma unroll
    for (int ft = 0; ft < 8; ++ft) {
      v16bf b = lds_load_frag(Bb + (ft * 16) * LDK);
      acc[ft] = __builtin_amdgcn_wmma_f32_16x16x32_bf16(
          false, a, false, b, (short)0, acc[ft], false, false);
    }
    if (more) commit((it + 1) & 1);   // write other buffer; readers passed barrier
  }

  const int m0 = w * 16;
  const int rowadd = (lane >> 4) << 3;
  const int col = lane & 15;
#pragma unroll
  for (int ft = 0; ft < 8; ++ft) {
#pragma unroll
    for (int j = 0; j < 8; ++j) {
      const int e = e0 + m0 + j + rowadd;
      if (e < E_) {
        float v = acc[ft][j] * de[(size_t)r * E_ + e];
        tX[((size_t)r * E_ + e) * F_ + ft * 16 + col] = (__bf16)v;
      }
    }
  }
}

// ---------------- GEMM2: Hn[r,n,f] = dv_is[r,n] * sum_e H[r,n,e]*tX[r,e,f] --
__global__ void __launch_bounds__(256)
k_gemm2(const float* __restrict__ H, const __bf16* __restrict__ tX,
        const float* __restrict__ dv, __bf16* __restrict__ Hn) {
  __shared__ alignas(16) __bf16 Abuf[2][128 * LDK];   // [n_local][e_local]
  __shared__ alignas(16) __bf16 Bbuf[2][128 * LDK];   // [f][e_local]
  const int r   = blockIdx.y;
  const int nb0 = blockIdx.x * 128;
  const int t   = threadIdx.x;
  const int w   = t >> 5;
  const int lane = t & 31;
  const int rowA = t >> 1;          // 0..127  (A: no transpose)
  const int cA   = (t & 1) * 16;    // 0 or 16
  const int r0 = (t >> 5) * 4;      // B rows (e_local)
  const int c0 = (t & 31) * 4;      // B cols (f)

  const size_t Hr = (size_t)r * N_ * E_;
  const bool n_full = (nb0 + 128 <= N_);

  v8f acc[8];
  v8f zero = {};
#pragma unroll
  for (int i = 0; i < 8; ++i) acc[i] = zero;

  float Ar16[16];
  uint2 Br[4];

  auto stage = [&](int e0) {
    const bool e_full = (e0 + 32 <= E_);
    if (n_full && e_full) {
      const v4f* rp = reinterpret_cast<const v4f*>(
          H + Hr + (size_t)(nb0 + rowA) * E_ + e0 + cA);
#pragma unroll
      for (int q = 0; q < 4; ++q) {
        v4f v = __builtin_nontemporal_load(rp + q);
        Ar16[q * 4 + 0] = v[0]; Ar16[q * 4 + 1] = v[1];
        Ar16[q * 4 + 2] = v[2]; Ar16[q * 4 + 3] = v[3];
      }
    } else {
      const int n = nb0 + rowA;
      const bool nok = n < N_;
      const size_t ro = Hr + (size_t)imin(n, N_ - 1) * E_;
#pragma unroll
      for (int i = 0; i < 16; ++i) {
        const int e = e0 + cA + i;
        float v = H[ro + imin(e, E_ - 1)];
        Ar16[i] = (nok && e < E_) ? v : 0.f;
      }
    }
#pragma unroll
    for (int j = 0; j < 4; ++j) {
      const int e = e0 + r0 + j;
      if (e_full) {
        Br[j] = *reinterpret_cast<const uint2*>(
            tX + ((size_t)r * E_ + e) * F_ + c0);
      } else {
        uint2 b = *reinterpret_cast<const uint2*>(
            tX + ((size_t)r * E_ + imin(e, E_ - 1)) * F_ + c0);
        if (e >= E_) { b.x = 0u; b.y = 0u; }
        Br[j] = b;
      }
    }
  };
  auto commit = [&](int buf) {
    __bf16* A = &Abuf[buf][0];
#pragma unroll
    for (int q = 0; q < 2; ++q) {
      uint4 d;
      d.x = pk2(Ar16[q * 8 + 0], Ar16[q * 8 + 1]);
      d.y = pk2(Ar16[q * 8 + 2], Ar16[q * 8 + 3]);
      d.z = pk2(Ar16[q * 8 + 4], Ar16[q * 8 + 5]);
      d.w = pk2(Ar16[q * 8 + 6], Ar16[q * 8 + 7]);
      *reinterpret_cast<uint4*>(A + rowA * LDK + cA + q * 8) = d;
    }
    commit_tr4(&Bbuf[buf][0], c0, r0, Br);
  };

  const int STEPS = (E_ + 31) / 32;   // 63
  stage(0);
  commit(0);
  for (int it = 0; it < STEPS; ++it) {
    const bool more = (it + 1) < STEPS;
    if (more) stage((it + 1) * 32);
    __syncthreads();
    const __bf16* Ab = &Abuf[it & 1][0];
    const __bf16* Bb = &Bbuf[it & 1][0];
    v16bf a = lds_load_frag(Ab + (w * 16) * LDK);
#pragma unroll
    for (int ft = 0; ft < 8; ++ft) {
      v16bf b = lds_load_frag(Bb + (ft * 16) * LDK);
      acc[ft] = __builtin_amdgcn_wmma_f32_16x16x32_bf16(
          false, a, false, b, (short)0, acc[ft], false, false);
    }
    if (more) commit((it + 1) & 1);
  }

  const int m0 = w * 16;
  const int rowadd = (lane >> 4) << 3;
  const int col = lane & 15;
#pragma unroll
  for (int ft = 0; ft < 8; ++ft) {
#pragma unroll
    for (int j = 0; j < 8; ++j) {
      const int n = nb0 + m0 + j + rowadd;
      if (n < N_) {
        float v = acc[ft][j] * dv[(size_t)r * N_ + n];
        Hn[((size_t)r * N_ + n) * F_ + ft * 16 + col] = (__bf16)v;
      }
    }
  }
}

// --------- GEMM3: out[n,o] = sum_r gate_r * Hn[r] @ W[r] + bias -------------
__global__ void __launch_bounds__(256)
k_gemm3(const __bf16* __restrict__ Hn, const float* __restrict__ W,
        const float* __restrict__ ri, const float* __restrict__ bias,
        float* __restrict__ out) {
  __shared__ alignas(16) __bf16 Abuf[2][128 * LDK];   // [n_local][k]
  __shared__ alignas(16) __bf16 Bbuf[2][128 * LDK];   // [o][k]
  const int nb0 = blockIdx.x * 128;
  const int t   = threadIdx.x;
  const int w   = t >> 5;
  const int lane = t & 31;
  const int rowA = t >> 1;
  const int cA   = (t & 1) * 16;
  const int r0 = (t >> 5) * 4;      // B rows (k)
  const int c0 = (t & 31) * 4;      // B cols (o)

  const bool n_full = (nb0 + 128 <= N_);

  float gg[R_];
#pragma unroll
  for (int r = 0; r < R_; ++r) gg[r] = 1.f / (1.f + expf(-ri[r]));

  v8f acc[8];
  v8f zero = {};
#pragma unroll
  for (int i = 0; i < 8; ++i) acc[i] = zero;

  uint4 Ar4[2];
  float Wr[4][4];

  auto stage = [&](int s) {
    const int r  = s >> 2;
    const int k0 = (s & 3) * 32;
    const int n = nb0 + rowA;
    const size_t ro = ((size_t)r * N_ + imin(n, N_ - 1)) * F_ + k0 + cA;
    const uint4* ap = reinterpret_cast<const uint4*>(Hn + ro);
    uint4 v0 = ap[0];
    uint4 v1 = ap[1];
    if (!n_full && n >= N_) {
      v0.x = v0.y = v0.z = v0.w = 0u;
      v1.x = v1.y = v1.z = v1.w = 0u;
    }
    Ar4[0] = v0;
    Ar4[1] = v1;
    const float g = gg[r];
#pragma unroll
    for (int j = 0; j < 4; ++j) {
      const float4* rp = reinterpret_cast<const float4*>(
          W + ((size_t)r * F_ + k0 + r0 + j) * O_ + c0);
      float4 v = *rp;
      Wr[j][0] = v.x * g; Wr[j][1] = v.y * g;
      Wr[j][2] = v.z * g; Wr[j][3] = v.w * g;
    }
  };
  auto commit = [&](int buf) {
    uint4* A = reinterpret_cast<uint4*>(&Abuf[buf][0] + rowA * LDK + cA);
    A[0] = Ar4[0];
    A[1] = Ar4[1];
    commit_tr4f(&Bbuf[buf][0], c0, r0, Wr);
  };

  const int STEPS = R_ * (F_ / 32);   // 16
  stage(0);
  commit(0);
  for (int it = 0; it < STEPS; ++it) {
    const bool more = (it + 1) < STEPS;
    if (more) stage(it + 1);
    __syncthreads();
    const __bf16* Ab = &Abuf[it & 1][0];
    const __bf16* Bb = &Bbuf[it & 1][0];
    v16bf a = lds_load_frag(Ab + (w * 16) * LDK);
#pragma unroll
    for (int ft = 0; ft < 8; ++ft) {
      v16bf b = lds_load_frag(Bb + (ft * 16) * LDK);
      acc[ft] = __builtin_amdgcn_wmma_f32_16x16x32_bf16(
          false, a, false, b, (short)0, acc[ft], false, false);
    }
    if (more) commit((it + 1) & 1);
  }

  const int m0 = w * 16;
  const int rowadd = (lane >> 4) << 3;
  const int col = lane & 15;
#pragma unroll
  for (int ft = 0; ft < 8; ++ft) {
    const int o = ft * 16 + col;
    const float b = bias[o];
#pragma unroll
    for (int j = 0; j < 8; ++j) {
      const int n = nb0 + m0 + j + rowadd;
      if (n < N_) out[(size_t)n * O_ + o] = acc[ft][j] + b;
    }
  }
}

// ---------------- launcher --------------------------------------------------
extern "C" void kernel_launch(void* const* d_in, const int* in_sizes, int n_in,
                              void* d_out, int out_size, void* d_ws, size_t ws_size,
                              hipStream_t stream) {
  const float* X    = (const float*)d_in[0];   // [N,F]
  const float* H    = (const float*)d_in[1];   // [R,N,E]
  const float* W    = (const float*)d_in[2];   // [R,F,O]
  const float* ri   = (const float*)d_in[3];   // [R]
  const float* bias = (const float*)d_in[4];   // [O]
  float* out = (float*)d_out;                  // [N,O]

  // Workspace layout (bytes):
  //   dv  f32 [R,N]      @ 0
  //   de  f32 [R,E]      @ 160,000
  //   Xs  bf16 [R,N,F]   @ 192,000
  //   tX  bf16 [R,E,F]   @ 10,432,000
  //   Hn  bf16 [R,N,F]   @ 12,480,000   (total ~22.7 MB)
  char* ws = (char*)d_ws;
  float*  dv = (float*)(ws + 0);
  float*  de = (float*)(ws + 160000);
  __bf16* Xs = (__bf16*)(ws + 192000);
  __bf16* tX = (__bf16*)(ws + 10432000);
  __bf16* Hn = (__bf16*)(ws + 12480000);

  (void)hipMemsetAsync(de, 0, (size_t)R_ * E_ * sizeof(float), stream);

  k_dv     <<<dim3((N_ + 7) / 8, R_),            256, 0, stream>>>(H, dv);
  k_de_part<<<dim3((E_ + 255) / 256, 40, R_),    256, 0, stream>>>(H, de);
  k_de_fin <<<(R_ * E_ + 255) / 256,             256, 0, stream>>>(de);
  {
    size_t tot = (size_t)R_ * N_ * F_;
    k_scaleX<<<(unsigned)((tot + 255) / 256),    256, 0, stream>>>(X, dv, Xs);
  }
  k_gemm1  <<<dim3((E_ + 127) / 128, R_),        256, 0, stream>>>(H, Xs, de, tX);
  k_gemm2  <<<dim3((N_ + 127) / 128, R_),        256, 0, stream>>>(H, tX, dv, Hn);
  k_gemm3  <<<dim3((N_ + 127) / 128),            256, 0, stream>>>(Hn, W, ri, bias, out);
}
